// ChempropEncoder_79267916415369
// MI455X (gfx1250) — matrix operand order repaired and checked by reading, not compiled, not run on hardware
//
#include <hip/hip_runtime.h>

typedef __attribute__((ext_vector_type(16))) _Float16 v16h;
typedef __attribute__((ext_vector_type(8)))  _Float16 v8h;
typedef __attribute__((ext_vector_type(8)))  float    v8f;

#define NA 131072     // atoms
#define NB 262144     // bonds
#define NM 4096       // molecules
#define DV 72
#define DE 14
#define DH 300
#define EMBD 256

// K paddings (multiple of 32)
#define KP_I 96       // 72+14=86 -> 96
#define KP_H 320      // 300 -> 320
#define KP_O 384      // 72+300=372 -> 384
#define KP_E 256      // 256
// N (row) paddings of transposed weight buffers: 2 * NTH * 16
#define NP_DH 320     // 300 -> 2*10*16
#define NP_EM 256     // 256 -> 2*8*16

// ---------------------------------------------------------------------------
// LDS-staged WMMA GEMM:
//   D[M,Nvalid] = act( A(f16,[M,Kpad]) * Bt(f16,[2*NTH*16, Kpad])^T (+Cadd)(+bias) )
// Optional STORE2: D gets the raw result, D2 gets relu(result)  (fused H0 / H).
// Block = 256 threads (8 waves) covering 64 rows x all N tiles.
//   wave = wrow (0..3: 16-row strip) x whalf (0..1: column half of NTH tiles).
// A strip (64 x 32) is double-buffered in LDS via global_load_async_to_lds_b128,
// overlapped with WMMA on the previous K stage.  A is read from HBM exactly once.
// ---------------------------------------------------------------------------
template <int NTH, bool RELU, bool ADDC, bool BIAS, bool STORE2>
__global__ __launch_bounds__(256)
void gemm64_wmma(const _Float16* __restrict__ A,
                 const _Float16* __restrict__ Bt,
                 const float* __restrict__ Cadd,
                 const float* __restrict__ bias,
                 float* __restrict__ D,
                 float* __restrict__ D2,
                 int Kpad, int Nvalid) {
    __shared__ _Float16 sA[2][64 * 32];   // 2 x 4 KB double buffer

    const int tid   = threadIdx.x;
    const int lane  = tid & 31;
    const int wave  = tid >> 5;
    const int wrow  = wave >> 1;          // 0..3
    const int whalf = wave & 1;           // 0..1
    const long long m0 = (long long)blockIdx.x * 64;

    // ---- async staging: each thread copies one 16-byte chunk per stage ----
    const int srow = tid >> 2;            // 0..63 : row within 64-row strip
    const int sseg = tid & 3;             // 0..3  : 16B segment within 64B row
    const _Float16* gsrc = A + (m0 + srow) * (long long)Kpad + sseg * 8;

    auto issue = [&](int stage, int k0) {
        unsigned lds = (unsigned)(unsigned long long)(&sA[stage][srow * 32 + sseg * 8]);
        unsigned long long ga = (unsigned long long)(gsrc + k0);
        asm volatile("global_load_async_to_lds_b128 %0, %1, off"
                     :: "v"(lds), "v"(ga) : "memory");
    };

    v8f acc[NTH];
#pragma unroll
    for (int nt = 0; nt < NTH; ++nt) acc[nt] = (v8f){0.f,0.f,0.f,0.f,0.f,0.f,0.f,0.f};

    const int mrow  = lane & 15;
    const int khalf = (lane >> 4) << 4;   // 0/16 (A K sub-offset within 32)
    const int kbb   = (lane >> 4) << 4;   // 0/16 (B K sub-offset within 32)
    // B fragment base: column n = lane&15 of this wave's first tile
    const _Float16* bbase =
        Bt + (long long)(whalf * NTH * 16 + (lane & 15)) * Kpad + kbb;

    const int nstages = Kpad >> 5;
    issue(0, 0);
    for (int s = 0; s < nstages; ++s) {
        const int k0 = s << 5;
        asm volatile("s_wait_asynccnt 0" ::: "memory");
        __syncthreads();
        if (s + 1 < nstages) issue((s + 1) & 1, k0 + 32);   // overlap next stage

        const _Float16* ar = &sA[s & 1][(wrow * 16 + mrow) * 32 + khalf];
        v8h alo = *(const v8h*)ar;             // ds_load_b128
        v8h ahi = *(const v8h*)(ar + 8);       // ds_load_b128
        v16h afrag = __builtin_shufflevector(alo, ahi,
                         0, 1, 2, 3, 4, 5, 6, 7, 8, 9, 10, 11, 12, 13, 14, 15);
        __builtin_prefetch(bbase + k0 + 32, 0, 2);          // weights stay hot
#pragma unroll
        for (int nt = 0; nt < NTH; ++nt) {
            v16h bfrag = *(const v16h*)(bbase + (long long)nt * 16 * Kpad + k0);
            acc[nt] = __builtin_amdgcn_wmma_f32_16x16x32_f16(
                          false, afrag, false, bfrag,
                          (short)0, acc[nt], false, false);
        }
    }

    // ---- epilogue ----
    const int colbase = whalf * NTH * 16 + (lane & 15);
    const long long rbase = m0 + wrow * 16 + ((lane >> 4) << 3);
#pragma unroll
    for (int nt = 0; nt < NTH; ++nt) {
        const int col = colbase + nt * 16;
        if (col < Nvalid) {
            const float bv = BIAS ? bias[col] : 0.0f;
#pragma unroll
            for (int r = 0; r < 8; ++r) {
                long long idx = (rbase + r) * Nvalid + col;
                float v = acc[nt][r];
                if (ADDC) v += Cadd[idx];
                if (BIAS) v += bv;
                if (RELU) v = fmaxf(v, 0.0f);
                D[idx] = v;
                if (STORE2) D2[idx] = fmaxf(v, 0.0f);   // fused relu copy
            }
        }
    }
}

// ---------------------------------------------------------------------------
// Elementwise / pack / scatter kernels
// ---------------------------------------------------------------------------
__global__ void k_zero_f32(float* __restrict__ p, long long n) {
    long long i = (long long)blockIdx.x * blockDim.x + threadIdx.x;
    long long st = (long long)gridDim.x * blockDim.x;
    for (; i < n; i += st) p[i] = 0.0f;
}

// Bt[n][k] = (k<K && n<N) ? W[k*N+n] : 0   (weight transpose + f16 + pad)
__global__ void k_pack_wt(const float* __restrict__ W, _Float16* __restrict__ Bt,
                          int K, int N, int Kpad, long long total) {
    long long i = (long long)blockIdx.x * blockDim.x + threadIdx.x;
    if (i >= total) return;
    int k = (int)(i % Kpad);
    int n = (int)(i / Kpad);
    float v = (k < K && n < N) ? W[(long long)k * N + n] : 0.0f;
    Bt[i] = (_Float16)v;
}

// X16[b][k] = k<72 ? V[src[b]][k] : k<86 ? E[b][k-72] : 0
__global__ void k_build_X(const float* __restrict__ V, const float* __restrict__ E,
                          const int* __restrict__ src, _Float16* __restrict__ X) {
    long long i = (long long)blockIdx.x * blockDim.x + threadIdx.x;
    if (i >= (long long)NB * KP_I) return;
    int k = (int)(i % KP_I);
    long long b = i / KP_I;
    float v = 0.0f;
    if (k < DV)            v = V[(long long)src[b] * DV + k];
    else if (k < DV + DE)  v = E[b * DE + (k - DV)];
    X[i] = (_Float16)v;
}

// Mv[dst[b]][c] += H[b][c]
__global__ void k_segsum(const float* __restrict__ H, const int* __restrict__ dst,
                         float* __restrict__ Mv) {
    long long i = (long long)blockIdx.x * blockDim.x + threadIdx.x;
    if (i >= (long long)NB * DH) return;
    int c = (int)(i % DH);
    long long b = i / DH;
    atomicAdd(&Mv[(long long)dst[b] * DH + c], H[i]);
}

// M16[b][k] = k<300 ? Mv[src[b]][k] - H[rev[b]][k] : 0
__global__ void k_build_M(const float* __restrict__ Mv, const float* __restrict__ H,
                          const int* __restrict__ src, const int* __restrict__ rev,
                          _Float16* __restrict__ M16) {
    long long i = (long long)blockIdx.x * blockDim.x + threadIdx.x;
    if (i >= (long long)NB * KP_H) return;
    int k = (int)(i % KP_H);
    long long b = i / KP_H;
    float v = 0.0f;
    if (k < DH) v = Mv[(long long)src[b] * DH + k] - H[(long long)rev[b] * DH + k];
    M16[i] = (_Float16)v;
}

// Xo16[a][k] = k<72 ? V[a][k] : k<372 ? Mv[a][k-72] : 0
__global__ void k_build_Xo(const float* __restrict__ V, const float* __restrict__ Mv,
                           _Float16* __restrict__ Xo) {
    long long i = (long long)blockIdx.x * blockDim.x + threadIdx.x;
    if (i >= (long long)NA * KP_O) return;
    int k = (int)(i % KP_O);
    long long a = i / KP_O;
    float v = 0.0f;
    if (k < DV)           v = V[a * DV + k];
    else if (k < DV + DH) v = Mv[a * DH + (k - DV)];
    Xo[i] = (_Float16)v;
}

// sums[batch[a]][c] += Hv[a][c];  counts[batch[a]] += 1 (once per atom)
__global__ void k_mol_accum(const float* __restrict__ Hv, const int* __restrict__ batch,
                            float* __restrict__ sums, float* __restrict__ counts) {
    long long i = (long long)blockIdx.x * blockDim.x + threadIdx.x;
    if (i >= (long long)NA * DH) return;
    int c = (int)(i % DH);
    long long a = i / DH;
    int m = batch[a];
    atomicAdd(&sums[(long long)m * DH + c], Hv[i]);
    if (c == 0) atomicAdd(&counts[m], 1.0f);
}

// Z16[m][k] = k<300 ? sums[m][k]/max(counts[m],1) : 0
__global__ void k_mean_cvt(const float* __restrict__ sums, const float* __restrict__ counts,
                           _Float16* __restrict__ Z16) {
    long long i = (long long)blockIdx.x * blockDim.x + threadIdx.x;
    if (i >= (long long)NM * KP_H) return;
    int k = (int)(i % KP_H);
    long long m = i / KP_H;
    float v = 0.0f;
    if (k < DH) v = sums[m * DH + k] / fmaxf(counts[m], 1.0f);
    Z16[i] = (_Float16)v;
}

__global__ void k_cvt_f16(const float* __restrict__ a, _Float16* __restrict__ o, long long n) {
    long long i = (long long)blockIdx.x * blockDim.x + threadIdx.x;
    long long st = (long long)gridDim.x * blockDim.x;
    for (; i < n; i += st) o[i] = (_Float16)a[i];
}

// ---------------------------------------------------------------------------
// Launcher
// ---------------------------------------------------------------------------
static inline long long cdiv(long long a, long long b) { return (a + b - 1) / b; }

extern "C" void kernel_launch(void* const* d_in, const int* in_sizes, int n_in,
                              void* d_out, int out_size, void* d_ws, size_t ws_size,
                              hipStream_t stream) {
    (void)in_sizes; (void)n_in; (void)out_size; (void)ws_size;
    const float* V    = (const float*)d_in[0];
    const float* E    = (const float*)d_in[1];
    const int*   src  = (const int*)d_in[2];
    const int*   dst  = (const int*)d_in[3];
    const int*   rev  = (const int*)d_in[4];
    const int*   batch= (const int*)d_in[5];
    const float* W_i  = (const float*)d_in[7];
    const float* W_h  = (const float*)d_in[8];
    const float* W_o  = (const float*)d_in[9];
    const float* b_o  = (const float*)d_in[10];
    const float* W1   = (const float*)d_in[11];
    const float* b1   = (const float*)d_in[12];
    const float* W2   = (const float*)d_in[13];
    const float* b2   = (const float*)d_in[14];
    float* out = (float*)d_out;

    // --- workspace carve (256-byte aligned slices) ---
    char* ws = (char*)d_ws;
    size_t off = 0;
    auto carve = [&](size_t bytes) { char* p = ws + off; off += (bytes + 255) & ~(size_t)255; return p; };
    _Float16* X16   = (_Float16*)carve((size_t)NB * KP_I * 2);
    float*    H0    = (float*)   carve((size_t)NB * DH   * 4);
    float*    H     = (float*)   carve((size_t)NB * DH   * 4);
    _Float16* M16   = (_Float16*)carve((size_t)NB * KP_H * 2);
    float*    Mv    = (float*)   carve((size_t)NA * DH   * 4);
    _Float16* Xo16  = (_Float16*)carve((size_t)NA * KP_O * 2);
    float*    Hv    = (float*)   carve((size_t)NA * DH   * 4);
    float*    sums  = (float*)   carve((size_t)NM * DH   * 4);
    float*    cnts  = (float*)   carve((size_t)NM * 4);
    _Float16* Z16   = (_Float16*)carve((size_t)NM * KP_H * 2);
    float*    z1    = (float*)   carve((size_t)NM * EMBD * 4);
    _Float16* z116  = (_Float16*)carve((size_t)NM * EMBD * 2);
    _Float16* Wi_t  = (_Float16*)carve((size_t)NP_DH * KP_I * 2);
    _Float16* Wh_t  = (_Float16*)carve((size_t)NP_DH * KP_H * 2);
    _Float16* Wo_t  = (_Float16*)carve((size_t)NP_DH * KP_O * 2);
    _Float16* W1_t  = (_Float16*)carve((size_t)NP_EM * KP_H * 2);
    _Float16* W2_t  = (_Float16*)carve((size_t)NP_EM * KP_E * 2);

    const int T = 256;
    // --- pack weights (f32 -> f16, transposed, zero-padded) ---
    {
        long long t;
        t = (long long)NP_DH * KP_I;
        k_pack_wt<<<cdiv(t, T), T, 0, stream>>>(W_i, Wi_t, DV + DE, DH, KP_I, t);
        t = (long long)NP_DH * KP_H;
        k_pack_wt<<<cdiv(t, T), T, 0, stream>>>(W_h, Wh_t, DH, DH, KP_H, t);
        t = (long long)NP_DH * KP_O;
        k_pack_wt<<<cdiv(t, T), T, 0, stream>>>(W_o, Wo_t, DV + DH, DH, KP_O, t);
        t = (long long)NP_EM * KP_H;
        k_pack_wt<<<cdiv(t, T), T, 0, stream>>>(W1, W1_t, DH, EMBD, KP_H, t);
        t = (long long)NP_EM * KP_E;
        k_pack_wt<<<cdiv(t, T), T, 0, stream>>>(W2, W2_t, EMBD, EMBD, KP_E, t);
    }

    // --- H0 = [V[src]|E] @ W_i ; H = relu(H0)  (fused dual store) ---
    k_build_X<<<cdiv((long long)NB * KP_I, T), T, 0, stream>>>(V, E, src, X16);
    gemm64_wmma<10, false, false, false, true><<<NB / 64, 256, 0, stream>>>(
        X16, Wi_t, nullptr, nullptr, H0, H, KP_I, DH);

    // --- message passing: DEPTH-1 = 2 iterations ---
    for (int it = 0; it < 2; ++it) {
        k_zero_f32<<<2048, T, 0, stream>>>(Mv, (long long)NA * DH);
        k_segsum<<<cdiv((long long)NB * DH, T), T, 0, stream>>>(H, dst, Mv);
        k_build_M<<<cdiv((long long)NB * KP_H, T), T, 0, stream>>>(Mv, H, src, rev, M16);
        gemm64_wmma<10, true, true, false, false><<<NB / 64, 256, 0, stream>>>(
            M16, Wh_t, H0, nullptr, H, nullptr, KP_H, DH);
    }

    // --- final aggregation to atoms: Hv = relu([V|Mv] @ W_o + b_o) ---
    k_zero_f32<<<2048, T, 0, stream>>>(Mv, (long long)NA * DH);
    k_segsum<<<cdiv((long long)NB * DH, T), T, 0, stream>>>(H, dst, Mv);
    k_build_Xo<<<cdiv((long long)NA * KP_O, T), T, 0, stream>>>(V, Mv, Xo16);
    gemm64_wmma<10, true, false, true, false><<<NA / 64, 256, 0, stream>>>(
        Xo16, Wo_t, nullptr, b_o, Hv, nullptr, KP_O, DH);

    // --- molecule mean pooling ---
    k_zero_f32<<<512, T, 0, stream>>>(sums, (long long)NM * DH);
    k_zero_f32<<<16, T, 0, stream>>>(cnts, (long long)NM);
    k_mol_accum<<<cdiv((long long)NA * DH, T), T, 0, stream>>>(Hv, batch, sums, cnts);
    k_mean_cvt<<<cdiv((long long)NM * KP_H, T), T, 0, stream>>>(sums, cnts, Z16);

    // --- projection MLP: out = relu(Z @ W1 + b1) @ W2 + b2 ---
    gemm64_wmma<8, true, false, true, false><<<NM / 64, 256, 0, stream>>>(
        Z16, W1_t, nullptr, b1, z1, nullptr, KP_H, EMBD);
    k_cvt_f16<<<512, T, 0, stream>>>(z1, z116, (long long)NM * EMBD);
    gemm64_wmma<8, false, false, true, false><<<NM / 64, 256, 0, stream>>>(
        z116, W2_t, nullptr, b2, out, nullptr, KP_E, EMBD);
}
